// MultiHeadAttention_72825465471427
// MI455X (gfx1250) — compile-verified, matrix operand order checked by reading, not run on hardware
//
#include <hip/hip_runtime.h>
#include <hip/hip_bf16.h>
#include <hip/hip_fp16.h>

// ---------------- problem constants ----------------
#define NB   2        // batch
#define NT   2048     // seq len
#define NE   1024     // embed
#define NH   16       // heads
#define ND   64       // head size
#define SCALE 0.125f  // 64^-0.5
#define NEG_BIG (-1.0e30f)

typedef __attribute__((ext_vector_type(16))) __bf16 v16bf;
typedef __attribute__((ext_vector_type(8)))  __bf16 v8bf;
typedef __attribute__((ext_vector_type(8)))  float  v8f;

union FragBF {
    v16bf v;
    v8bf  h[2];
};

// WMMA wrapper: D = A*B + C, bf16 inputs, f32 accum
__device__ __forceinline__ v8f wmma_bf16(const v16bf& a, const v16bf& b, const v8f& c) {
    return __builtin_amdgcn_wmma_f32_16x16x32_bf16(false, a, false, b, (short)0, c, false, false);
}

// Load a 16-element bf16 fragment: per-lane base pointer, elements {0..7} and {16..23}
// (kb offset already folded into ptr by caller). 16B-aligned.
__device__ __forceinline__ FragBF load_frag(const __bf16* p) {
    FragBF f;
    f.h[0] = *(const v8bf*)(p);
    f.h[1] = *(const v8bf*)(p + 16);
    return f;
}

// ---------------- prep kernels (one-time bf16 conversion / transpose) ----------------

__global__ void cvt_f32_to_bf16(const float* __restrict__ src, __bf16* __restrict__ dst, int n) {
    int i = blockIdx.x * blockDim.x + threadIdx.x;
    if (i < n) dst[i] = (__bf16)src[i];
}

// Wt[mat][h][d][e] = W{k,q,v}[h][e][d], bf16.  total = 3*H*D*E = 3 * 2^20
__global__ void cvt_w_transpose(const float* __restrict__ Wk, const float* __restrict__ Wq,
                                const float* __restrict__ Wv, __bf16* __restrict__ Wt) {
    int i = blockIdx.x * blockDim.x + threadIdx.x;
    int e   = i & (NE - 1);
    int d   = (i >> 10) & (ND - 1);
    int h   = (i >> 16) & (NH - 1);
    int mat = i >> 20;
    const float* W = (mat == 0) ? Wk : ((mat == 1) ? Wq : Wv);
    Wt[i] = (__bf16)W[((size_t)h * NE + e) * ND + d];
}

// ---------------- kernel 1: QKV projections ----------------
// One wave computes one 16(t) x 64(d) tile of one of {K,Q,V} for one (b,h).
// K,Q stored [B,H,T,D]; V stored transposed [B,H,D,T].
__global__ void proj_kernel(const __bf16* __restrict__ Xbf, const __bf16* __restrict__ Wt,
                            __bf16* __restrict__ Kb, __bf16* __restrict__ Qb,
                            __bf16* __restrict__ Vt) {
    int wid  = blockIdx.x * (blockDim.x >> 5) + (threadIdx.x >> 5);
    int lane = threadIdx.x & 31;
    int tt   = wid & 127;           // T/16 = 128 tiles
    int rest = wid >> 7;
    int h    = rest & (NH - 1); rest >>= 4;
    int b    = rest & 1;
    int mat  = rest >> 1;           // 0=K 1=Q 2=V
    int t0   = tt * 16;

    const __bf16* X = Xbf + (size_t)b * NT * NE;
    const __bf16* W = Wt + ((size_t)mat * NH + h) * ND * NE;

    int m     = lane & 15;
    int kb    = (lane < 16) ? 0 : 8;
    int rbase = (lane < 16) ? 0 : 8;

    v8f acc[4] = {};
    for (int e0 = 0; e0 < NE; e0 += 32) {
        FragBF a = load_frag(X + (size_t)(t0 + m) * NE + e0 + kb);
#pragma unroll
        for (int nt = 0; nt < 4; ++nt) {
            FragBF bf = load_frag(W + (size_t)(nt * 16 + m) * NE + e0 + kb);
            acc[nt] = wmma_bf16(a.v, bf.v, acc[nt]);
        }
    }

    if (mat == 2) {
        __bf16* out = Vt + ((size_t)b * NH + h) * ND * NT;
#pragma unroll
        for (int nt = 0; nt < 4; ++nt) {
            int d = nt * 16 + m;
#pragma unroll
            for (int r = 0; r < 8; ++r)
                out[(size_t)d * NT + t0 + rbase + r] = (__bf16)acc[nt][r];
        }
    } else {
        __bf16* out = ((mat == 0) ? Kb : Qb) + (((size_t)b * NH + h) * NT + t0) * ND;
#pragma unroll
        for (int nt = 0; nt < 4; ++nt)
#pragma unroll
            for (int r = 0; r < 8; ++r)
                out[(size_t)(rbase + r) * ND + nt * 16 + m] = (__bf16)acc[nt][r];
    }
}

// ---------------- kernel 2: attention (flash-style, wei = K @ Q^T) ----------------
// One wave owns a 16-row t-block of one (b,h). 8 waves per block.
__global__ void attn_kernel(const __bf16* __restrict__ Kb, const __bf16* __restrict__ Qb,
                            const __bf16* __restrict__ Vt, __bf16* __restrict__ OutCat) {
    __shared__ __bf16 pbuf[8 * 16 * 32];   // 1KB per wave: P tile 16x32 bf16

    int wid  = blockIdx.x * (blockDim.x >> 5) + (threadIdx.x >> 5);
    int lane = threadIdx.x & 31;
    int tt   = wid & 127;
    int h    = (wid >> 7) & (NH - 1);
    int b    = wid >> 11;
    int t0   = tt * 16;

    const __bf16* Kp = Kb + (((size_t)b * NH + h) * NT) * ND;
    const __bf16* Qp = Qb + (((size_t)b * NH + h) * NT) * ND;
    const __bf16* Vp = Vt + ((size_t)b * NH + h) * ND * NT;
    __bf16* pl = pbuf + (threadIdx.x >> 5) * (16 * 32);

    int m     = lane & 15;
    int kb    = (lane < 16) ? 0 : 8;
    int rbase = (lane < 16) ? 0 : 8;

    // K-block A fragments (rows t0..t0+15, d split 0..31 / 32..63) held all loop
    const __bf16* kp = Kp + (size_t)(t0 + m) * ND + kb;
    FragBF ka0 = load_frag(kp);
    FragBF ka1 = load_frag(kp + 32);

    v8f oacc[4] = {};
    float rm[8], rl[8];
#pragma unroll
    for (int r = 0; r < 8; ++r) { rm[r] = NEG_BIG; rl[r] = 0.0f; }

    for (int s0 = 0; s0 <= t0; s0 += 32) {
        // ---- scores for two 16-wide s halves ----
        float P[2][8];
        float tmax[8];
#pragma unroll
        for (int r = 0; r < 8; ++r) tmax[r] = NEG_BIG;

#pragma unroll
        for (int half = 0; half < 2; ++half) {
            int sbase = s0 + half * 16;
            const __bf16* qp = Qp + (size_t)(sbase + m) * ND + kb;  // column n=lane&15 -> s=sbase+n
            FragBF qb0 = load_frag(qp);
            FragBF qb1 = load_frag(qp + 32);
            v8f c = {};
            c = wmma_bf16(ka0.v, qb0.v, c);
            c = wmma_bf16(ka1.v, qb1.v, c);
            int sg = sbase + m;
#pragma unroll
            for (int r = 0; r < 8; ++r) {
                int tg = t0 + rbase + r;
                float x = (sg <= tg) ? (c[r] * SCALE) : NEG_BIG;
                P[half][r] = x;
                tmax[r] = fmaxf(tmax[r], x);
            }
        }
        // ---- row max across the 16 lanes holding each row ----
#pragma unroll
        for (int off = 1; off < 16; off <<= 1)
#pragma unroll
            for (int r = 0; r < 8; ++r)
                tmax[r] = fmaxf(tmax[r], __shfl_xor(tmax[r], off, 32));

        float tsum[8];
#pragma unroll
        for (int r = 0; r < 8; ++r) {
            float newm = fmaxf(rm[r], tmax[r]);
            float corr = __expf(rm[r] - newm);
            rm[r] = newm;
            float p0 = __expf(P[0][r] - newm);
            float p1 = __expf(P[1][r] - newm);
            P[0][r] = p0; P[1][r] = p1;
            tsum[r] = p0 + p1;
            rl[r] *= corr;
#pragma unroll
            for (int nt = 0; nt < 4; ++nt) oacc[nt][r] *= corr;
        }
#pragma unroll
        for (int off = 1; off < 16; off <<= 1)
#pragma unroll
            for (int r = 0; r < 8; ++r)
                tsum[r] += __shfl_xor(tsum[r], off, 32);
#pragma unroll
        for (int r = 0; r < 8; ++r) rl[r] += tsum[r];

        // ---- C-layout -> A-layout bounce through LDS (intra-wave, in-order DS) ----
#pragma unroll
        for (int half = 0; half < 2; ++half)
#pragma unroll
            for (int r = 0; r < 8; ++r)
                pl[(rbase + r) * 32 + half * 16 + m] = (__bf16)P[half][r];
        asm volatile("s_wait_dscnt 0" ::: "memory");

        FragBF pa = load_frag(pl + m * 32 + kb);

        // ---- out += P(16x32) @ V(32x64): B frags contiguous from V^T ----
#pragma unroll
        for (int nt = 0; nt < 4; ++nt) {
            FragBF vb = load_frag(Vp + (size_t)(nt * 16 + m) * NT + s0 + kb);
            oacc[nt] = wmma_bf16(pa.v, vb.v, oacc[nt]);
        }
        asm volatile("s_wait_dscnt 0" ::: "memory");
    }

    // ---- epilogue: normalize and store concatenated heads [B,T,H*D] bf16 ----
    __bf16* op = OutCat + ((size_t)b * NT + t0) * NE + h * ND;
#pragma unroll
    for (int nt = 0; nt < 4; ++nt)
#pragma unroll
        for (int r = 0; r < 8; ++r)
            op[(size_t)(rbase + r) * NE + nt * 16 + m] = (__bf16)(oacc[nt][r] / rl[r]);
}

// ---------------- kernel 3: z = outcat @ Wo^T + bo, f32 output ----------------
__global__ void zproj_kernel(const __bf16* __restrict__ OutCat, const __bf16* __restrict__ Wob,
                             const float* __restrict__ bo, float* __restrict__ Z) {
    int wid  = blockIdx.x * (blockDim.x >> 5) + (threadIdx.x >> 5);
    int lane = threadIdx.x & 31;
    int jt   = wid & 15;            // E/64 = 16 column tiles
    int tt   = (wid >> 4) & 127;
    int b    = wid >> 11;
    int j0   = jt * 64;
    int t0   = tt * 16;

    int m     = lane & 15;
    int kb    = (lane < 16) ? 0 : 8;
    int rbase = (lane < 16) ? 0 : 8;

    const __bf16* X = OutCat + (size_t)b * NT * NE;

    v8f acc[4] = {};
    for (int e0 = 0; e0 < NE; e0 += 32) {
        FragBF a = load_frag(X + (size_t)(t0 + m) * NE + e0 + kb);
#pragma unroll
        for (int nt = 0; nt < 4; ++nt) {
            // B[k,n] = Wo[j0+nt*16+n, e0+k] : contiguous along e (row-major Wo)
            FragBF bf = load_frag(Wob + (size_t)(j0 + nt * 16 + m) * NE + e0 + kb);
            acc[nt] = wmma_bf16(a.v, bf.v, acc[nt]);
        }
    }
#pragma unroll
    for (int nt = 0; nt < 4; ++nt) {
        float bias = bo[j0 + nt * 16 + m];
#pragma unroll
        for (int r = 0; r < 8; ++r)
            Z[((size_t)b * NT + t0 + rbase + r) * NE + j0 + nt * 16 + m] = acc[nt][r] + bias;
    }
}

// ---------------- host launch ----------------
extern "C" void kernel_launch(void* const* d_in, const int* in_sizes, int n_in,
                              void* d_out, int out_size, void* d_ws, size_t ws_size,
                              hipStream_t stream) {
    const float* x  = (const float*)d_in[0];   // [B,T,E]
    const float* Wk = (const float*)d_in[1];   // [H,E,D]
    const float* Wq = (const float*)d_in[2];
    const float* Wv = (const float*)d_in[3];
    const float* Wo = (const float*)d_in[4];   // [E,E]
    const float* bo = (const float*)d_in[5];   // [E]
    float* z = (float*)d_out;                  // [B,T,E] f32

    // workspace layout (bf16 elements)
    __bf16* ws = (__bf16*)d_ws;
    const size_t nXE  = (size_t)NB * NT * NE;          // 4,194,304
    const size_t nWT  = (size_t)3 * NH * ND * NE;      // 3,145,728
    const size_t nWO  = (size_t)NE * NE;               // 1,048,576
    const size_t nKQV = (size_t)NB * NH * NT * ND;     // 4,194,304
    __bf16* Xbf    = ws;
    __bf16* Wt     = Xbf + nXE;
    __bf16* Wob    = Wt + nWT;
    __bf16* Kb     = Wob + nWO;
    __bf16* Qb     = Kb + nKQV;
    __bf16* Vt     = Qb + nKQV;
    __bf16* OutCat = Vt + nKQV;

    const int TPB = 256;

    // prep: bf16 conversions / transposes
    cvt_f32_to_bf16<<<(int)(nXE / TPB), TPB, 0, stream>>>(x, Xbf, (int)nXE);
    cvt_w_transpose<<<(int)(nWT / TPB), TPB, 0, stream>>>(Wk, Wq, Wv, Wt);
    cvt_f32_to_bf16<<<(int)(nWO / TPB), TPB, 0, stream>>>(Wo, Wob, (int)nWO);

    // QKV projections: 3*B*H*(T/16) = 12288 wave tasks, 8 waves/block
    proj_kernel<<<12288 / 8, TPB, 0, stream>>>(Xbf, Wt, Kb, Qb, Vt);

    // attention: B*H*(T/16) = 4096 wave tasks
    attn_kernel<<<4096 / 8, TPB, 0, stream>>>(Kb, Qb, Vt, OutCat);

    // output projection: B*(T/16)*(E/64) = 4096 wave tasks
    zproj_kernel<<<4096 / 8, TPB, 0, stream>>>(OutCat, Wob, bo, z);
}
